// ParallelTransformerBlock_6287832121667
// MI455X (gfx1250) — matrix operand order repaired and checked by reading, not compiled, not run on hardware
//
#include <hip/hip_runtime.h>
#include <hip/hip_bf16.h>

// ---------------------------------------------------------------------------
// MI455X (gfx1250) parallel transformer block, bf16 WMMA pipeline:
//   1. to_bf16_T     : W1,W2 fp32 -> bf16, transposed to (N x K)
//   2. layernorm_k   : LN(x) -> bf16 activations
//   3. gemm_bf16<1>  : [8192x1024]@[1024x6144] WMMA GEMM; K-tiles staged by
//                      the Tensor Data Mover (tensor_load_to_lds, TENSORcnt),
//                      double buffered; epilogue: qkv bf16 + gelu -> concat
//   4. qkprep        : head RMSNorm + RoPE (+0.125 prescale on q), v transposed
//   5. attn_k        : flash attention; QK^T / PV on v_wmma_f32_16x16x32_bf16;
//                      K/V staged with global_load_async_to_lds_b128 (ASYNCcnt,
//                      double buffered); softmax reductions on DPP16
//   6. gemm_bf16<2>  : [8192x4096]@[4096x1024] WMMA GEMM + bias + residual
// ---------------------------------------------------------------------------

typedef __attribute__((ext_vector_type(16))) __bf16 v16bf;
typedef __attribute__((ext_vector_type(8)))  float  v8f;
typedef unsigned int u32;
typedef __attribute__((ext_vector_type(4))) u32 u32x4;
typedef __attribute__((ext_vector_type(8))) int i32x8;
typedef __attribute__((ext_vector_type(4))) int i32x4;

#define DIM   1024
#define NHEAD 16
#define HDIM  64
#define SEQ   2048
#define NTOK  8192      // B*L
#define N1    6144      // 3*DIM + MLP
#define K1    1024
#define N2    1024
#define K2    4096      // DIM + MLP

union Frag {
  v16bf v;
  uint4 q[2];
  __bf16 h[16];
};

__device__ __forceinline__ v8f vzero() {
  v8f z = {0.f, 0.f, 0.f, 0.f, 0.f, 0.f, 0.f, 0.f};
  return z;
}

// ---- CDNA5 async memory->LDS (ASYNCcnt) ------------------------------------
// LDS operand = low 32 bits of the generic shared address (ISA 10.2: LDS
// aperture addresses truncate to the wave-relative LDS offset).
__device__ __forceinline__ void async_b128(__bf16* lds_dst, const __bf16* gsrc) {
  asm volatile("global_load_async_to_lds_b128 %0, %1, off"
               :: "v"((unsigned)(unsigned long long)lds_dst), "v"(gsrc)
               : "memory");
}
__device__ __forceinline__ void wait_async() {
  asm volatile("s_wait_asynccnt 0x0" ::: "memory");
}

// ---- CDNA5 Tensor Data Mover: 2D tile, global -> LDS (TENSORcnt) -----------
// D# bitfields per ISA 8.3-8.5: group0 = {count=1 | lds_addr | global_addr |
// type=2}; group1 = {data_size=1 (2B), tensor_dim0/1 = tile dims,
// tile_dim0/1, tensor_dim0_stride}. Tile rows packed row-major into LDS.
#if __has_builtin(__builtin_amdgcn_tensor_load_to_lds)
#define HAVE_TDM 1
__device__ __forceinline__ void tdm_load_2d(u32 lds_addr, const void* gaddr,
                                            u32 tile_x, u32 tile_y,
                                            u32 stride_x /*elems*/) {
  unsigned long long ga = (unsigned long long)gaddr;
  u32x4 g0;
  g0[0] = 1u;                                            // count=1, user mode
  g0[1] = lds_addr;                                      // lds_addr [63:32]
  g0[2] = (u32)(ga & 0xffffffffu);                       // global_addr[31:0]
  g0[3] = (u32)((ga >> 32) & 0x1ffffffu) | (2u << 30);   // addr[56:32]|type=2
  i32x8 g1;
  g1[0] = (int)(1u << 16);                               // data_size = 2 bytes
  g1[1] = (int)((tile_x & 0xffffu) << 16);               // tensor_dim0[15:0]
  g1[2] = (int)(((tile_x >> 16) & 0xffffu) |             // tensor_dim0[31:16]
                ((tile_y & 0xffffu) << 16));             // tensor_dim1[15:0]
  g1[3] = (int)(((tile_y >> 16) & 0xffffu) |             // tensor_dim1[31:16]
                ((tile_x & 0xffffu) << 16));             // tile_dim0
  g1[4] = (int)(tile_y & 0xffffu);                       // tile_dim1 (dim2=0)
  g1[5] = (int)stride_x;                                 // dim0_stride[31:0]
  g1[6] = 0;
  g1[7] = 0;
  i32x4 z4 = {0, 0, 0, 0};
#if defined(__clang_major__) && __clang_major__ >= 23
  i32x8 z8 = {0, 0, 0, 0, 0, 0, 0, 0};
  __builtin_amdgcn_tensor_load_to_lds(g0, g1, z4, z4, z8, 0);
#else
  __builtin_amdgcn_tensor_load_to_lds(g0, g1, z4, z4, 0);
#endif
}
#endif

__device__ __forceinline__ v8f wmma_bf16(v16bf a, v16bf b, v8f c) {
  return __builtin_amdgcn_wmma_f32_16x16x32_bf16(false, a, false, b,
                                                 (short)0, c, false, false);
}

// ---- DPP16 butterfly reductions over a 16-lane half (VALU-only) ------------
__device__ __forceinline__ float red16_max(float x) {
  union { float f; int i; } a, b;
  a.f = x;
  b.i = __builtin_amdgcn_update_dpp(a.i, a.i, 0xB1, 0xF, 0xF, true);
  a.f = fmaxf(a.f, b.f);
  b.i = __builtin_amdgcn_update_dpp(a.i, a.i, 0x4E, 0xF, 0xF, true);
  a.f = fmaxf(a.f, b.f);
  b.i = __builtin_amdgcn_update_dpp(a.i, a.i, 0x141, 0xF, 0xF, true);
  a.f = fmaxf(a.f, b.f);
  b.i = __builtin_amdgcn_update_dpp(a.i, a.i, 0x140, 0xF, 0xF, true);
  return fmaxf(a.f, b.f);
}
__device__ __forceinline__ float red16_sum(float x) {
  union { float f; int i; } a, b;
  a.f = x;
  b.i = __builtin_amdgcn_update_dpp(a.i, a.i, 0xB1, 0xF, 0xF, true);
  a.f = a.f + b.f;
  b.i = __builtin_amdgcn_update_dpp(a.i, a.i, 0x4E, 0xF, 0xF, true);
  a.f = a.f + b.f;
  b.i = __builtin_amdgcn_update_dpp(a.i, a.i, 0x141, 0xF, 0xF, true);
  a.f = a.f + b.f;
  b.i = __builtin_amdgcn_update_dpp(a.i, a.i, 0x140, 0xF, 0xF, true);
  return a.f + b.f;
}

__device__ __forceinline__ float gelu_tanh(float x) {
  float x3 = x * x * x;
  return 0.5f * x * (1.0f + tanhf(0.7978845608028654f * (x + 0.044715f * x3)));
}

// --------------------------- weight convert + transpose ---------------------
__global__ void to_bf16_T(const float* __restrict__ src, __bf16* __restrict__ dst,
                          int K, int N) {
  size_t idx = (size_t)blockIdx.x * 256 + threadIdx.x;
  int n = (int)(idx / (size_t)K);
  int k = (int)(idx % (size_t)K);
  dst[idx] = (__bf16)src[(size_t)k * N + n];
}

// --------------------------- layernorm --------------------------------------
__global__ void layernorm_k(const float* __restrict__ x, __bf16* __restrict__ xn) {
  __shared__ float r1[256];
  __shared__ float r2[256];
  const int row = blockIdx.x, tid = threadIdx.x;
  const float* xr = x + (size_t)row * DIM;
  float v[4], s = 0.f, ss = 0.f;
#pragma unroll
  for (int i = 0; i < 4; ++i) {
    v[i] = xr[tid + i * 256];
    s += v[i];
    ss += v[i] * v[i];
  }
  r1[tid] = s; r2[tid] = ss;
  __syncthreads();
  for (int st = 128; st > 0; st >>= 1) {
    if (tid < st) { r1[tid] += r1[tid + st]; r2[tid] += r2[tid + st]; }
    __syncthreads();
  }
  float mu  = r1[0] * (1.f / DIM);
  float var = r2[0] * (1.f / DIM) - mu * mu;
  float rs  = rsqrtf(var + 1e-6f);
  __bf16* o = xn + (size_t)row * DIM;
#pragma unroll
  for (int i = 0; i < 4; ++i) o[tid + i * 256] = (__bf16)((v[i] - mu) * rs);
}

// --------------------------- WMMA GEMM --------------------------------------
// A: (M x K) bf16 row-major.  BT: (N x K) bf16 row-major.
// 256 threads = 8 waves; block tile 128x128; wave tile 64(M) x 32(N).
// K-tiles staged by the Tensor Data Mover (one descriptor per tile, issued by
// wave 0; EXEC-ignored per-wave op), double buffered on TENSORcnt.
template <int MODE>
__global__ __launch_bounds__(256)
void gemm_bf16(const __bf16* __restrict__ A, const __bf16* __restrict__ BT,
               const float* __restrict__ bias, const float* __restrict__ resid,
               __bf16* __restrict__ qkv_out, __bf16* __restrict__ concat_out,
               float* __restrict__ f32_out, int N, int K) {
  __shared__ __align__(16) __bf16 As[2][128 * 32];
  __shared__ __align__(16) __bf16 Bs[2][128 * 32];

  const int tid  = threadIdx.x;
  const int lane = tid & 31;
  const int wave = tid >> 5;
  const int wm   = wave & 1;
  const int wn   = wave >> 1;
  const int l16  = lane & 15;
  const int lh   = lane >> 4;
  const int mBase = blockIdx.y * 128;
  const int nBase = blockIdx.x * 128;

  v8f acc[4][2];
#pragma unroll
  for (int i = 0; i < 4; ++i)
#pragma unroll
    for (int j = 0; j < 2; ++j) acc[i][j] = vzero();

  const __bf16* aTile = A  + (size_t)mBase * K;
  const __bf16* bTile = BT + (size_t)nBase * K;

#if HAVE_TDM
  const u32 asLds = (u32)(unsigned long long)&As[0][0];
  const u32 bsLds = (u32)(unsigned long long)&Bs[0][0];
  auto stage = [&](int k0, int buf) {
    tdm_load_2d(asLds + (u32)buf * (128 * 32 * 2), aTile + k0, 32, 128, (u32)K);
    tdm_load_2d(bsLds + (u32)buf * (128 * 32 * 2), bTile + k0, 32, 128, (u32)K);
  };
  if (wave == 0) stage(0, 0);
#else
  const int sr = tid >> 1;
  const int sc = (tid & 1) * 16;
  auto stage_async = [&](int k0, int buf) {
    async_b128(&As[buf][sr * 32 + sc],     aTile + (size_t)sr * K + sc + k0);
    async_b128(&As[buf][sr * 32 + sc + 8], aTile + (size_t)sr * K + sc + k0 + 8);
    async_b128(&Bs[buf][sr * 32 + sc],     bTile + (size_t)sr * K + sc + k0);
    async_b128(&Bs[buf][sr * 32 + sc + 8], bTile + (size_t)sr * K + sc + k0 + 8);
  };
  stage_async(0, 0);
#endif

  int cur = 0;
  for (int k0 = 0; k0 < K; k0 += 32) {
#if HAVE_TDM
    if (wave == 0) __builtin_amdgcn_s_wait_tensorcnt(0);
    __syncthreads();
    if (k0 + 32 < K && wave == 0) stage(k0 + 32, cur ^ 1);
#else
    wait_async();
    __syncthreads();
    if (k0 + 32 < K) stage_async(k0 + 32, cur ^ 1);
#endif
    if (k0 + 64 < K) {  // deep lookahead into L2 (global_prefetch_b8)
      __builtin_prefetch(aTile + (size_t)(tid >> 1) * K + k0 + 64, 0, 0);
      __builtin_prefetch(bTile + (size_t)(tid >> 1) * K + k0 + 64, 0, 0);
    }

    Frag af[4], bf[2];
#pragma unroll
    for (int i = 0; i < 4; ++i) {
      const __bf16* p = &As[cur][(wm * 64 + i * 16 + l16) * 32 + lh * 8];
      af[i].q[0] = *(const uint4*)p;
      af[i].q[1] = *(const uint4*)(p + 16);
    }
#pragma unroll
    for (int j = 0; j < 2; ++j) {
      const __bf16* p = &Bs[cur][(wn * 32 + j * 16 + l16) * 32 + lh * 8];
      bf[j].q[0] = *(const uint4*)p;
      bf[j].q[1] = *(const uint4*)(p + 16);
    }
#pragma unroll
    for (int i = 0; i < 4; ++i)
#pragma unroll
      for (int j = 0; j < 2; ++j)
        acc[i][j] = wmma_bf16(af[i].v, bf[j].v, acc[i][j]);
    cur ^= 1;
  }

  // epilogue: C layout = VGPR r -> row r + 8*(lane/16), col = lane%16
#pragma unroll
  for (int i = 0; i < 4; ++i)
#pragma unroll
    for (int j = 0; j < 2; ++j) {
      const int col = nBase + wn * 32 + j * 16 + l16;
      const float bb = bias[col];
#pragma unroll
      for (int r = 0; r < 8; ++r) {
        const int row = mBase + wm * 64 + i * 16 + r + 8 * lh;
        float c = acc[i][j][r] + bb;
        if constexpr (MODE == 1) {
          if (col < 3072)
            qkv_out[(size_t)row * 3072 + col] = (__bf16)c;
          else
            concat_out[(size_t)row * 4096 + 1024 + (col - 3072)] =
                (__bf16)gelu_tanh(c);
        } else {
          f32_out[(size_t)row * 1024 + col] =
              resid[(size_t)row * 1024 + col] + c;
        }
      }
    }
}

// --------------------------- qk rmsnorm + rope + v transpose ----------------
__global__ void qkprep(const __bf16* __restrict__ qkvb, const float* __restrict__ pe,
                       const float* __restrict__ qsc, const float* __restrict__ ksc,
                       __bf16* __restrict__ qr, __bf16* __restrict__ kr,
                       __bf16* __restrict__ vT) {
  const int lane = threadIdx.x & 31;
  const int wave = threadIdx.x >> 5;
  const int row = blockIdx.x * 8 + wave;  // (b*16+h)*2048 + l
  const int l  = row & (SEQ - 1);
  const int bh = row >> 11;
  const int h  = bh & 15;
  const int b  = bh >> 4;
  const int d0 = lane * 2, d1 = d0 + 1;

  const __bf16* src = qkvb + ((size_t)(b * SEQ + l)) * 3072 + h * HDIM;
  float q0 = (float)src[d0],        q1 = (float)src[d1];
  float k0 = (float)src[1024 + d0], k1 = (float)src[1024 + d1];
  float v0 = (float)src[2048 + d0], v1 = (float)src[2048 + d1];

  float sq = q0 * q0 + q1 * q1;
  float sk = k0 * k0 + k1 * k1;
#pragma unroll
  for (int m = 1; m < 32; m <<= 1) { sq += __shfl_xor(sq, m); sk += __shfl_xor(sk, m); }
  const float rq = rsqrtf(sq * (1.f / HDIM) + 1e-6f);
  const float rk = rsqrtf(sk * (1.f / HDIM) + 1e-6f);
  q0 = q0 * rq * qsc[d0] * 0.125f;
  q1 = q1 * rq * qsc[d1] * 0.125f;
  k0 = k0 * rk * ksc[d0];
  k1 = k1 * rk * ksc[d1];

  const float* p = pe + ((size_t)l * 32 + lane) * 4;  // pe[l, i, j, c]
  const float p00 = p[0], p01 = p[1], p10 = p[2], p11 = p[3];
  const float qo0 = p00 * q0 + p01 * q1, qo1 = p10 * q0 + p11 * q1;
  const float ko0 = p00 * k0 + p01 * k1, ko1 = p10 * k0 + p11 * k1;

  __bf16* qd = qr + (size_t)row * HDIM;
  __bf16* kd = kr + (size_t)row * HDIM;
  qd[d0] = (__bf16)qo0; qd[d1] = (__bf16)qo1;
  kd[d0] = (__bf16)ko0; kd[d1] = (__bf16)ko1;
  vT[((size_t)bh * HDIM + d0) * SEQ + l] = (__bf16)v0;
  vT[((size_t)bh * HDIM + d1) * SEQ + l] = (__bf16)v1;
}

// --------------------------- flash attention --------------------------------
__global__ __launch_bounds__(128)
void attn_k(const __bf16* __restrict__ qr, const __bf16* __restrict__ kr,
            const __bf16* __restrict__ vT, __bf16* __restrict__ concat) {
  __shared__ __align__(16) __bf16 Ks[2][32 * HDIM];   // [key][d]
  __shared__ __align__(16) __bf16 Vs[2][HDIM * 32];   // [d][key]
  __shared__ __align__(16) __bf16 Pw[4][16 * 32];     // per-wave P bounce

  const int tid  = threadIdx.x;
  const int lane = tid & 31;
  const int wave = tid >> 5;
  const int l16  = lane & 15;
  const int lh   = lane >> 4;
  const int bh    = blockIdx.x >> 5;
  const int qBase = (blockIdx.x & 31) * 64;

  Frag qf[2];
  const __bf16* qrow = qr + ((size_t)bh * SEQ + qBase + wave * 16 + l16) * HDIM;
#pragma unroll
  for (int kb = 0; kb < 2; ++kb) {
    const int db = kb * 32 + lh * 8;
    qf[kb].q[0] = *(const uint4*)(qrow + db);
    qf[kb].q[1] = *(const uint4*)(qrow + db + 16);
  }

  const int key0 = tid >> 2, dd0 = (tid & 3) * 16;
  const int dv0  = tid >> 1, kk0 = (tid & 1) * 16;
  const __bf16* kSrc = kr + ((size_t)bh * SEQ + key0) * HDIM + dd0;
  const __bf16* vSrc = vT + ((size_t)bh * HDIM + dv0) * SEQ + kk0;
  auto stageKV = [&](int kc, int buf) {
    async_b128(&Ks[buf][key0 * HDIM + dd0],     kSrc + (size_t)kc * HDIM);
    async_b128(&Ks[buf][key0 * HDIM + dd0 + 8], kSrc + (size_t)kc * HDIM + 8);
    async_b128(&Vs[buf][dv0 * 32 + kk0],        vSrc + kc);
    async_b128(&Vs[buf][dv0 * 32 + kk0 + 8],    vSrc + kc + 8);
  };

  float m8[8], l8[8], alpha[8];
  v8f o[4];
#pragma unroll
  for (int r = 0; r < 8; ++r) { m8[r] = -1e30f; l8[r] = 0.f; }
#pragma unroll
  for (int t = 0; t < 4; ++t) o[t] = vzero();

  stageKV(0, 0);
  int cur = 0;
  for (int kc = 0; kc < SEQ; kc += 32) {
    wait_async();
    __syncthreads();
    if (kc + 32 < SEQ) stageKV(kc + 32, cur ^ 1);

    // ---- S = Q K^T for two 16-key tiles ----
    Frag kf;
    v8f s0 = vzero(), s1 = vzero();
#pragma unroll
    for (int kb = 0; kb < 2; ++kb) {
      const int db = kb * 32 + lh * 8;
      const __bf16* p0 = &Ks[cur][(0 * 16 + l16) * HDIM + db];
      kf.q[0] = *(const uint4*)p0; kf.q[1] = *(const uint4*)(p0 + 16);
      s0 = wmma_bf16(qf[kb].v, kf.v, s0);
      const __bf16* p1 = &Ks[cur][(1 * 16 + l16) * HDIM + db];
      kf.q[0] = *(const uint4*)p1; kf.q[1] = *(const uint4*)(p1 + 16);
      s1 = wmma_bf16(qf[kb].v, kf.v, s1);
    }

    // ---- online softmax (row = r + 8*lh); DPP16 reductions ----
#pragma unroll
    for (int r = 0; r < 8; ++r) {
      const float mx = red16_max(fmaxf(s0[r], s1[r]));
      const float mnew = fmaxf(m8[r], mx);
      const float a = __expf(m8[r] - mnew);
      const float p0 = __expf(s0[r] - mnew);
      const float p1 = __expf(s1[r] - mnew);
      const float ps = red16_sum(p0 + p1);
      l8[r] = l8[r] * a + ps;
      m8[r] = mnew;
      alpha[r] = a;
      const int prow = r + 8 * lh;
      Pw[wave][prow * 32 + l16]      = (__bf16)p0;
      Pw[wave][prow * 32 + 16 + l16] = (__bf16)p1;
    }
#pragma unroll
    for (int t = 0; t < 4; ++t)
#pragma unroll
      for (int r = 0; r < 8; ++r) o[t][r] = o[t][r] * alpha[r];

    // ---- O += P V (P via LDS bounce; same-wave LDS is in-order) ----
    Frag pf;
    const __bf16* pp = &Pw[wave][l16 * 32 + lh * 8];
    pf.q[0] = *(const uint4*)pp;
    pf.q[1] = *(const uint4*)(pp + 16);
#pragma unroll
    for (int t = 0; t < 4; ++t) {
      Frag vf;
      const __bf16* vb = &Vs[cur][(t * 16 + l16) * 32 + lh * 8];
      vf.q[0] = *(const uint4*)vb; vf.q[1] = *(const uint4*)(vb + 16);
      o[t] = wmma_bf16(pf.v, vf.v, o[t]);
    }
    cur ^= 1;
  }

  const int b = bh >> 4, h = bh & 15;
#pragma unroll
  for (int r = 0; r < 8; ++r) {
    const float inv = 1.0f / l8[r];
    const int row = qBase + wave * 16 + r + 8 * lh;
    __bf16* orow = concat + ((size_t)(b * SEQ + row)) * 4096 + h * HDIM;
#pragma unroll
    for (int t = 0; t < 4; ++t) orow[t * 16 + l16] = (__bf16)(o[t][r] * inv);
  }
}

// --------------------------- host launch ------------------------------------
extern "C" void kernel_launch(void* const* d_in, const int* in_sizes, int n_in,
                              void* d_out, int out_size, void* d_ws, size_t ws_size,
                              hipStream_t stream) {
  const float* x   = (const float*)d_in[0];
  const float* pe  = (const float*)d_in[1];
  const float* W1  = (const float*)d_in[2];
  const float* b1  = (const float*)d_in[3];
  const float* W2  = (const float*)d_in[4];
  const float* b2  = (const float*)d_in[5];
  const float* qsc = (const float*)d_in[6];
  const float* ksc = (const float*)d_in[7];
  float* out = (float*)d_out;
  (void)in_sizes; (void)n_in; (void)out_size; (void)ws_size;

  char* w = (char*)d_ws;
  size_t off = 0;
  auto carve = [&](size_t elems) -> __bf16* {
    __bf16* p = (__bf16*)(w + off);
    off += ((elems * sizeof(__bf16)) + 255) & ~(size_t)255;
    return p;
  };
  __bf16* W1T  = carve((size_t)N1 * K1);          // 12 MB
  __bf16* W2T  = carve((size_t)N2 * K2);          //  8 MB
  __bf16* xnb  = carve((size_t)NTOK * DIM);       // 16 MB
  __bf16* qkvb = carve((size_t)NTOK * 3072);      // 48 MB
  __bf16* cat  = carve((size_t)NTOK * 4096);      // 64 MB
  __bf16* qr   = carve((size_t)64 * SEQ * HDIM);  // 16 MB
  __bf16* kr   = carve((size_t)64 * SEQ * HDIM);  // 16 MB
  __bf16* vT   = carve((size_t)64 * SEQ * HDIM);  // 16 MB

  to_bf16_T<<<(N1 * K1) / 256, 256, 0, stream>>>(W1, W1T, K1, N1);
  to_bf16_T<<<(N2 * K2) / 256, 256, 0, stream>>>(W2, W2T, K2, N2);
  layernorm_k<<<NTOK, 256, 0, stream>>>(x, xnb);
  gemm_bf16<1><<<dim3(N1 / 128, NTOK / 128), 256, 0, stream>>>(
      xnb, W1T, b1, nullptr, qkvb, cat, nullptr, N1, K1);
  qkprep<<<(64 * SEQ) / 8, 256, 0, stream>>>(qkvb, pe, qsc, ksc, qr, kr, vT);
  attn_k<<<64 * (SEQ / 64), 128, 0, stream>>>(qr, kr, vT, cat);
  gemm_bf16<2><<<dim3(N2 / 128, NTOK / 128), 256, 0, stream>>>(
      cat, W2T, b2, x, nullptr, nullptr, out, N2, K2);
}
